// MoELayer_28381143892386
// MI455X (gfx1250) — matrix operand order repaired and checked by reading, not compiled, not run on hardware
//
#include <hip/hip_runtime.h>
#include <hip/hip_bf16.h>

// MoE SwiGLU top-4 router, CDNA5 (gfx1250) wave32 + WMMA bf16 + async-LDS staging.
// B=4,S=2048,D=256,F=1024,E=32,K=4; T=8192 tokens.

#define T_TOK 8192
#define DDIM  256
#define FDIM  1024
#define NEXP  32
#define CAP   8192   // worst-case tokens per expert
#define MT    32     // token-tile rows per workgroup
#define XP    264    // LDS pitch for x tile (bf16 elems), padded vs bank conflicts
#define HP    1032   // LDS pitch for h tile

typedef __bf16 bf16;
typedef __attribute__((ext_vector_type(16))) __bf16 v16bf;
typedef __attribute__((ext_vector_type(8)))  float  v8f;

union Frag { v16bf v; uint4 q[2]; };

#define WMMA_BF16(A, B, C) \
  __builtin_amdgcn_wmma_f32_16x16x32_bf16(false, (A), false, (B), (short)0, (C), false, false)

// ---------------- f32 -> bf16 conversion ----------------
__global__ void cvt_kernel(const float* __restrict__ s, bf16* __restrict__ d, int n) {
  int i = (blockIdx.x * blockDim.x + threadIdx.x) * 4;
  if (i + 3 < n) {
    float4 v = *(const float4*)(s + i);
    d[i + 0] = (bf16)v.x; d[i + 1] = (bf16)v.y;
    d[i + 2] = (bf16)v.z; d[i + 3] = (bf16)v.w;
  }
}

// ---------------- router: logits, top-4 softmax, scatter, aux stats ----------------
// Gather entries are encoded as (token << 2) | k so expert outputs can be
// written atomically-free into a per-(token,k) slot buffer.
__global__ void router_kernel(const float* __restrict__ x, const float* __restrict__ rw,
                              int* __restrict__ cnt, float* __restrict__ probSum,
                              int* __restrict__ tokIdx, float* __restrict__ tokWt) {
  __shared__ float sProb[NEXP];
  const int lane = threadIdx.x & 31;        // lane == expert id
  const int wv   = threadIdx.x >> 5;
  if (threadIdx.x < NEXP) sProb[threadIdx.x] = 0.f;
  __syncthreads();

  const int t = blockIdx.x * 4 + wv;        // one wave per token
  const float* xr = x  + (size_t)t   * DDIM;
  const float* wr = rw + (size_t)lane * DDIM;
  float acc = 0.f;
  #pragma unroll 4
  for (int d = 0; d < DDIM; d += 4) {
    float4 xv = *(const float4*)(xr + d);
    float4 wv4 = *(const float4*)(wr + d);
    acc += xv.x * wv4.x + xv.y * wv4.y + xv.z * wv4.z + xv.w * wv4.w;
  }

  // top-4 across the 32 lanes (tie-break by lower lane, matches top_k order)
  float lw = acc;
  float v0, v1, v2, v3; int i0, i1, i2, i3;
  #pragma unroll
  for (int k = 0; k < 4; k++) {
    float m = lw; int mi = lane;
    #pragma unroll
    for (int off = 16; off > 0; off >>= 1) {
      float om = __shfl_xor(m, off, 32);
      int   oi = __shfl_xor(mi, off, 32);
      if (om > m || (om == m && oi < mi)) { m = om; mi = oi; }
    }
    if (k == 0) { v0 = m; i0 = mi; } else if (k == 1) { v1 = m; i1 = mi; }
    else if (k == 2) { v2 = m; i2 = mi; } else { v3 = m; i3 = mi; }
    if (lane == mi) lw = -3.4e38f;
  }
  // softmax over the 4 selected logits (uniform across wave)
  float e0 = __expf(v0 - v0), e1 = __expf(v1 - v0), e2 = __expf(v2 - v0), e3 = __expf(v3 - v0);
  float inv4 = __builtin_amdgcn_rcpf(e0 + e1 + e2 + e3);
  // full softmax for aux loss (per-lane prob for its expert)
  float p = __expf(acc - v0);
  float ps = p;
  #pragma unroll
  for (int off = 16; off > 0; off >>= 1) ps += __shfl_xor(ps, off, 32);
  p *= __builtin_amdgcn_rcpf(ps);
  atomicAdd(&sProb[lane], p);

  float wsel = (lane == 0) ? e0 : (lane == 1) ? e1 : (lane == 2) ? e2 : e3;
  int   esel = (lane == 0) ? i0 : (lane == 1) ? i1 : (lane == 2) ? i2 : i3;
  if (lane < 4) {
    int pos = atomicAdd(&cnt[esel], 1);
    tokIdx[esel * CAP + pos] = (t << 2) | lane;
    tokWt [esel * CAP + pos] = wsel * inv4;
  }
  __syncthreads();
  if (threadIdx.x < NEXP) atomicAdd(&probSum[threadIdx.x], sProb[threadIdx.x]);
}

// ---------------- fused SwiGLU expert FFN (WMMA bf16) ----------------
__global__ __launch_bounds__(256) void expert_kernel(
    const bf16* __restrict__ xb,  const bf16* __restrict__ wgb,
    const bf16* __restrict__ wub, const bf16* __restrict__ wdb,
    const int* __restrict__ cnt,  const int* __restrict__ tokIdx,
    const float* __restrict__ tokWt, float* __restrict__ slots) {
  __shared__ bf16 sx[MT * XP];   // gathered x tile
  __shared__ bf16 sh[MT * HP];   // h = silu(gate)*up, bf16
  __shared__ int   sTok[MT];     // slot row = (token<<2)|k
  __shared__ float sWt[MT];

  const int e    = blockIdx.y;
  const int n_e  = cnt[e];
  const int base = blockIdx.x * MT;
  if (base >= n_e) return;
  const int rows = min(MT, n_e - base);

  const int tid  = threadIdx.x;
  const int lane = tid & 31;
  const int wv   = tid >> 5;

  if (tid < MT) {
    if (tid < rows) {
      sTok[tid] = tokIdx[e * CAP + base + tid];
      sWt [tid] = tokWt [e * CAP + base + tid];
    } else { sTok[tid] = 0; sWt[tid] = 0.f; }
  }
  { // stage gathered x rows via CDNA5 async global->LDS copies (ASYNCcnt path)
    int row = tid >> 3, seg = tid & 7;               // 8 segs * 32 bf16 (64B) per row
    bf16* dst = sx + row * XP + seg * 32;
    if (row < rows) {
      int entry = tokIdx[e * CAP + base + row];
      const bf16* src = xb + (size_t)(entry >> 2) * DDIM + seg * 32;
      unsigned           ldsa = (unsigned)(uintptr_t)dst;       // flat->LDS: low 32 bits
      unsigned long long ga   = (unsigned long long)(uintptr_t)src;
      asm volatile("global_load_async_to_lds_b128 %0, %1, off\n\t"
                   "global_load_async_to_lds_b128 %0, %1, off offset:16\n\t"
                   "global_load_async_to_lds_b128 %0, %1, off offset:32\n\t"
                   "global_load_async_to_lds_b128 %0, %1, off offset:48"
                   :: "v"(ldsa), "v"(ga) : "memory");
    } else {                                          // zero-pad tail rows
      uint4 z = {0, 0, 0, 0};
      uint4* d4 = (uint4*)dst;
      d4[0] = z; d4[1] = z; d4[2] = z; d4[3] = z;
    }
    asm volatile("s_wait_asynccnt 0x0" ::: "memory");
  }
  __syncthreads();

  const int krow = lane & 15;
  const int ka = (lane >> 4) << 3;   // A-frag K half offset (0 / 8)
  const int kb = (lane >> 4) << 4;   // unused; B-frag also uses ka (0 / 8)

  // ---- Phase 1: h = silu(x Wg^T) * (x Wu^T), 64 N-tiles of F split over 8 waves
  const bf16* wgE = wgb + (size_t)e * FDIM * DDIM;
  const bf16* wuE = wub + (size_t)e * FDIM * DDIM;
  for (int nt = wv; nt < FDIM / 16; nt += 8) {
    v8f g0 = {0,0,0,0,0,0,0,0}, g1 = g0, u0 = g0, u1 = g0;
    const int n = nt * 16 + krow;
    const bf16* bg = wgE + (size_t)n * DDIM + ka;
    const bf16* bu = wuE + (size_t)n * DDIM + ka;
    #pragma unroll
    for (int kk = 0; kk < DDIM; kk += 32) {
      Frag a0, a1, fg, fu;
      const bf16* pa0 = sx + krow * XP + kk + ka;
      const bf16* pa1 = sx + (16 + krow) * XP + kk + ka;
      a0.q[0] = *(const uint4*)pa0;        a0.q[1] = *(const uint4*)(pa0 + 16);
      a1.q[0] = *(const uint4*)pa1;        a1.q[1] = *(const uint4*)(pa1 + 16);
      fg.q[0] = *(const uint4*)(bg + kk);  fg.q[1] = *(const uint4*)(bg + kk + 16);
      fu.q[0] = *(const uint4*)(bu + kk);  fu.q[1] = *(const uint4*)(bu + kk + 16);
      g0 = WMMA_BF16(a0.v, fg.v, g0);  g1 = WMMA_BF16(a1.v, fg.v, g1);
      u0 = WMMA_BF16(a0.v, fu.v, u0);  u1 = WMMA_BF16(a1.v, fu.v, u1);
    }
    #pragma unroll
    for (int r = 0; r < 8; r++) {
      int m = r + ka;                                   // C layout: M = r + 8*(lane>=16)
      float gv0 = g0[r];
      sh[m * HP + n] = (bf16)(gv0 * __builtin_amdgcn_rcpf(1.f + __expf(-gv0)) * u0[r]);
      float gv1 = g1[r];
      sh[(16 + m) * HP + n] = (bf16)(gv1 * __builtin_amdgcn_rcpf(1.f + __expf(-gv1)) * u1[r]);
    }
  }
  __syncthreads();

  // ---- Phase 2: slot_row = (h Wd^T) * weight, plain stores (no atomics)
  const bf16* wdE = wdb + (size_t)e * DDIM * FDIM;
  #pragma unroll
  for (int i = 0; i < 2; i++) {
    const int nt = wv * 2 + i;
    v8f c0 = {0,0,0,0,0,0,0,0}, c1 = c0;
    const int n = nt * 16 + krow;
    const bf16* bd = wdE + (size_t)n * FDIM + ka;
    #pragma unroll 4
    for (int kk = 0; kk < FDIM; kk += 32) {
      Frag a0, a1, fb;
      const bf16* pa0 = sh + krow * HP + kk + ka;
      const bf16* pa1 = sh + (16 + krow) * HP + kk + ka;
      a0.q[0] = *(const uint4*)pa0;        a0.q[1] = *(const uint4*)(pa0 + 16);
      a1.q[0] = *(const uint4*)pa1;        a1.q[1] = *(const uint4*)(pa1 + 16);
      fb.q[0] = *(const uint4*)(bd + kk);  fb.q[1] = *(const uint4*)(bd + kk + 16);
      c0 = WMMA_BF16(a0.v, fb.v, c0);
      c1 = WMMA_BF16(a1.v, fb.v, c1);
    }
    #pragma unroll
    for (int r = 0; r < 8; r++) {
      int m0 = r + ka;
      if (m0 < rows) slots[(size_t)sTok[m0] * DDIM + n] = c0[r] * sWt[m0];
      int m1 = 16 + r + ka;
      if (m1 < rows) slots[(size_t)sTok[m1] * DDIM + n] = c1[r] * sWt[m1];
    }
  }
}

// ---------------- combine: out[t] = sum_k slots[t*4+k]  ----------------
__global__ void reduce_kernel(const float* __restrict__ slots, float* __restrict__ out) {
  int i = (blockIdx.x * blockDim.x + threadIdx.x) * 4;  // over T*D
  if (i >= T_TOK * DDIM) return;
  int t = i >> 8, d = i & 255;
  const float* s0 = slots + ((size_t)(t * 4) << 8) + d;
  float4 a = *(const float4*)(s0);
  float4 b = *(const float4*)(s0 + 256);
  float4 c = *(const float4*)(s0 + 512);
  float4 e = *(const float4*)(s0 + 768);
  float4 r;
  r.x = a.x + b.x + c.x + e.x;
  r.y = a.y + b.y + c.y + e.y;
  r.z = a.z + b.z + c.z + e.z;
  r.w = a.w + b.w + c.w + e.w;
  *(float4*)(out + i) = r;
}

// ---------------- aux loss ----------------
__global__ void aux_kernel(const int* __restrict__ cnt, const float* __restrict__ probSum,
                           float* __restrict__ aux) {
  int lane = threadIdx.x;
  float v = ((float)cnt[lane] / (float)T_TOK) * (probSum[lane] / (float)T_TOK);
  #pragma unroll
  for (int off = 16; off > 0; off >>= 1) v += __shfl_xor(v, off, 32);
  if (lane == 0) *aux = (float)NEXP * v;
}

extern "C" void kernel_launch(void* const* d_in, const int* in_sizes, int n_in,
                              void* d_out, int out_size, void* d_ws, size_t ws_size,
                              hipStream_t stream) {
  (void)in_sizes; (void)n_in; (void)out_size; (void)ws_size;
  const float* x        = (const float*)d_in[0];
  const float* router_w = (const float*)d_in[1];
  const float* w_gate   = (const float*)d_in[2];
  const float* w_up     = (const float*)d_in[3];
  const float* w_down   = (const float*)d_in[4];
  float* out = (float*)d_out;

  char* ws = (char*)d_ws;
  size_t off = 0;
  auto alloc = [&](size_t bytes) { void* p = ws + off; off = (off + bytes + 255) & ~(size_t)255; return p; };
  bf16*  xb      = (bf16*)alloc((size_t)T_TOK * DDIM * 2);
  bf16*  wgb     = (bf16*)alloc((size_t)NEXP * FDIM * DDIM * 2);
  bf16*  wub     = (bf16*)alloc((size_t)NEXP * FDIM * DDIM * 2);
  bf16*  wdb     = (bf16*)alloc((size_t)NEXP * DDIM * FDIM * 2);
  int*   cnt     = (int*)alloc(NEXP * 4);
  float* probSum = (float*)alloc(NEXP * 4);
  int*   tokIdx  = (int*)alloc((size_t)NEXP * CAP * 4);
  float* tokWt   = (float*)alloc((size_t)NEXP * CAP * 4);
  float* slots   = (float*)alloc((size_t)T_TOK * 4 * DDIM * 4);  // [T*K, D]

  hipMemsetAsync(cnt, 0, NEXP * 4, stream);
  hipMemsetAsync(probSum, 0, NEXP * 4, stream);

  cvt_kernel<<<(T_TOK * DDIM / 4 + 255) / 256, 256, 0, stream>>>(x, xb, T_TOK * DDIM);
  const int wn = NEXP * FDIM * DDIM;
  cvt_kernel<<<(wn / 4 + 255) / 256, 256, 0, stream>>>(w_gate, wgb, wn);
  cvt_kernel<<<(wn / 4 + 255) / 256, 256, 0, stream>>>(w_up,   wub, wn);
  cvt_kernel<<<(wn / 4 + 255) / 256, 256, 0, stream>>>(w_down, wdb, wn);

  router_kernel<<<T_TOK / 4, 128, 0, stream>>>(x, router_w, cnt, probSum, tokIdx, tokWt);

  expert_kernel<<<dim3(T_TOK / MT, NEXP), 256, 0, stream>>>(xb, wgb, wub, wdb,
                                                            cnt, tokIdx, tokWt, slots);

  reduce_kernel<<<(T_TOK * DDIM / 4 + 255) / 256, 256, 0, stream>>>(slots, out);

  aux_kernel<<<1, 32, 0, stream>>>(cnt, probSum, out + (size_t)T_TOK * DDIM);
}
